// DecoderLayer_63814624084528
// MI455X (gfx1250) — matrix operand order repaired and checked by reading, not compiled, not run on hardware
//
#include <hip/hip_runtime.h>
#include <hip/hip_bf16.h>
#include <math.h>

// ---------------------------------------------------------------------------
// CDNA5 (gfx1250) bf16 WMMA decoder layer.
// B=2, S=2048, E=1024, H=16, HD=64, HIDDEN=4096.
// GEMMs: v_wmma_f32_16x16x32_bf16 with TDM (tensor_load_to_lds) double-buffered
// tile staging. Attention: flash-style with WMMA + LDS bias table.
// ---------------------------------------------------------------------------

typedef __attribute__((ext_vector_type(16))) __bf16 v16bf;
typedef __attribute__((ext_vector_type(8)))  float  v8f;
typedef unsigned int u32x4 __attribute__((ext_vector_type(4)));
typedef int          i32x8 __attribute__((ext_vector_type(8)));
typedef int          i32x4 __attribute__((ext_vector_type(4)));

#if defined(__AMDGCN__) && __has_builtin(__builtin_amdgcn_tensor_load_to_lds)
#define USE_TDM 1
#else
#define USE_TDM 0
#endif

union ABFrag {
    v16bf v;
    unsigned short s[16];
    uint4 q[2];
};

__device__ __forceinline__ unsigned short f2bf(float f) {
    unsigned int u = __float_as_uint(f);
    u += 0x7fffu + ((u >> 16) & 1u);          // round-to-nearest-even
    return (unsigned short)(u >> 16);
}

__device__ __forceinline__ v8f v8f_zero() {
    v8f z;
#pragma unroll
    for (int i = 0; i < 8; ++i) z[i] = 0.0f;
    return z;
}

__device__ __forceinline__ v8f wmma_bf16(const ABFrag& a, const ABFrag& b, v8f c) {
    return __builtin_amdgcn_wmma_f32_16x16x32_bf16(
        false, a.v, false, b.v, (short)0, c, false, false);
}

__device__ __forceinline__ unsigned lds_off(const void* p) {
    // generic LDS address: low 32 bits are the LDS byte offset
    return (unsigned)(unsigned long long)p;
}

#if USE_TDM
// Issue one TDM load of a 128-row x 32-col bf16 tile (row stride = tensor_k
// elements in memory) into LDS with a 16-byte pad after every 64-byte row,
// reproducing the 40-ushort padded LDS row stride the WMMA fragment loads use.
// This toolchain's builtin takes 6 args: (g0 v4u, g1 v8i, g2 v4i, g3 v4i,
// extra v8i, cpol i32); groups 2/3 and the extra group are zero for 2-D tiles.
__device__ __forceinline__ void tdm_load_tile(
    const unsigned short* gsrc, unsigned lds_byte, int tensor_k, int tensor_rows)
{
    const unsigned long long ga = (unsigned long long)gsrc;
    u32x4 g0;
    g0[0] = 1u;                                              // count=1, user mode
    g0[1] = lds_byte;                                        // lds_addr
    g0[2] = (unsigned)(ga & 0xffffffffull);                  // global_addr[31:0]
    g0[3] = (unsigned)((ga >> 32) & 0x1ffffffull) | (2u << 30); // [56:32] | type=2
    i32x8 g1;
    g1[0] = (1 << 16)      // data_size = 2 bytes
          | (1 << 20)      // pad_enable
          | (3 << 22)      // pad_interval: 16 DWORDs (one 64B row)
          | (3 << 25);     // pad_amount: 4 DWORDs (16B -> stride 80B)
    g1[1] = (tensor_k & 0xffff) << 16;                       // tensor_dim0[15:0]
    g1[2] = ((tensor_k >> 16) & 0xffff) | ((tensor_rows & 0xffff) << 16);
    g1[3] = ((tensor_rows >> 16) & 0xffff) | (32 << 16);     // tile_dim0 = 32
    g1[4] = 128;                                             // tile_dim1 = 128
    g1[5] = tensor_k;                                        // dim0 stride low32
    g1[6] = 0;
    g1[7] = 0;
    i32x4 z4;
    z4[0] = 0; z4[1] = 0; z4[2] = 0; z4[3] = 0;
    i32x8 z8;
#pragma unroll
    for (int i = 0; i < 8; ++i) z8[i] = 0;
    __builtin_amdgcn_tensor_load_to_lds(g0, g1, z4, z4, z8, 0);
}
#endif

// ---------------------------------------------------------------------------
// Weight prep: W fp32 [K,N] -> WT bf16 [N,K]  (64x64 LDS tile transpose)
// ---------------------------------------------------------------------------
__global__ __launch_bounds__(256) void transpose_bf16_kernel(
    const float* __restrict__ W, unsigned short* __restrict__ WT, int K, int N)
{
    __shared__ unsigned short t[64][65];
    const int tid = threadIdx.x;
    const int bx = blockIdx.x * 64;   // N
    const int by = blockIdx.y * 64;   // K
    const int c = tid & 63;
    const int rb = tid >> 6;          // 0..3
#pragma unroll
    for (int i = 0; i < 16; ++i) {
        const int r = rb + i * 4;     // k-local
        t[c][r] = f2bf(W[(size_t)(by + r) * N + bx + c]);
    }
    __syncthreads();
#pragma unroll
    for (int i = 0; i < 16; ++i) {
        const int r = rb + i * 4;     // n-local
        WT[(size_t)(bx + r) * K + by + c] = t[r][c];
    }
}

// ---------------------------------------------------------------------------
// RMSNorm: one block per row of 1024, fp32 in -> bf16 out
// ---------------------------------------------------------------------------
__global__ __launch_bounds__(256) void rmsnorm_kernel(
    const float* __restrict__ x, const float* __restrict__ wgt,
    unsigned short* __restrict__ out, float eps)
{
    const int row = blockIdx.x;
    const float* xr = x + (size_t)row * 1024;
    const int tid = threadIdx.x;
    float vals[4];
    float s = 0.0f;
#pragma unroll
    for (int j = 0; j < 4; ++j) {
        vals[j] = xr[tid + 256 * j];
        s += vals[j] * vals[j];
    }
#pragma unroll
    for (int m = 16; m >= 1; m >>= 1) s += __shfl_xor(s, m, 32);
    __shared__ float red[8];
    if ((tid & 31) == 0) red[tid >> 5] = s;
    __syncthreads();
    float tot = 0.0f;
#pragma unroll
    for (int j = 0; j < 8; ++j) tot += red[j];
    const float inv = rsqrtf(tot * (1.0f / 1024.0f) + eps);
#pragma unroll
    for (int j = 0; j < 4; ++j) {
        const int i = tid + 256 * j;
        out[(size_t)row * 1024 + i] = f2bf(vals[j] * inv * wgt[i]);
    }
}

// ---------------------------------------------------------------------------
// bf16 WMMA GEMM: C[M,N] = A(bf16,[M,K]) * BT(bf16,[N,K])^T
// Block 256 = 8 waves; tile 128x128x32; wave tile 64x32 = 4x2 WMMA.
// Tiles staged to LDS by the Tensor Data Mover (double-buffered, wave 0
// issues, s_wait_tensorcnt overlaps DMA with WMMA), fallback: vector copies.
// EPI: 0 = fp32 store; 1 = fp32 store + residual; 2 = bf16 relu store
// ---------------------------------------------------------------------------
#define BM 128
#define BN 128
#define BK 32
#define LTS 40   // LDS row stride in ushorts (32 data + 8 pad = 80B)

template<int EPI>
__global__ __launch_bounds__(256) void gemm_bf16(
    const unsigned short* __restrict__ A,
    const unsigned short* __restrict__ BT,
    float* __restrict__ outF,
    unsigned short* __restrict__ outB,
    const float* __restrict__ R,
    int M, int N, int K)
{
    __shared__ __align__(16) unsigned short tiles[4][BM * LTS]; // A0,A1,B0,B1

    const int tid  = threadIdx.x;
    const int bm   = blockIdx.y * BM;
    const int bn   = blockIdx.x * BN;
    const int w    = tid >> 5;
    const int lane = tid & 31;
    const int lm   = lane & 15;
    const int hi   = lane >> 4;
    const int wm   = (w >> 2) * 64;
    const int wn   = (w & 3) * 32;

    const unsigned short* Abase = A  + (size_t)bm * K;
    const unsigned short* Bbase = BT + (size_t)bn * K;

    v8f acc[4][2];
#pragma unroll
    for (int i = 0; i < 4; ++i)
#pragma unroll
        for (int j = 0; j < 2; ++j) acc[i][j] = v8f_zero();

#if USE_TDM
    const unsigned ldsA0 = lds_off(&tiles[0][0]);
    const unsigned ldsA1 = lds_off(&tiles[1][0]);
    const unsigned ldsB0 = lds_off(&tiles[2][0]);
    const unsigned ldsB1 = lds_off(&tiles[3][0]);
    if (tid < 32) {
        tdm_load_tile(Abase, ldsA0, K, M);
        tdm_load_tile(Bbase, ldsB0, K, N);
    }
#endif

    const int nIter = K / BK;
    int buf = 0;
    for (int it = 0; it < nIter; ++it) {
        const int kk = it * BK;
#if USE_TDM
        if (tid < 32) {
            if (it + 1 < nIter) {
                // prefetch next tiles into the other buffer, then wait for
                // the two oldest (current buffer) DMAs only
                tdm_load_tile(Abase + kk + BK, buf ? ldsA0 : ldsA1, K, M);
                tdm_load_tile(Bbase + kk + BK, buf ? ldsB0 : ldsB1, K, N);
                __builtin_amdgcn_s_wait_tensorcnt(2);
            } else {
                __builtin_amdgcn_s_wait_tensorcnt(0);
            }
        }
#else
        {
            const int row  = tid >> 1;
            const int colh = (tid & 1) * 16;
            const uint4* sa = (const uint4*)(Abase + (size_t)row * K + kk + colh);
            uint4* da = (uint4*)&tiles[buf][row * LTS + colh];
            da[0] = sa[0]; da[1] = sa[1];
            const uint4* sb = (const uint4*)(Bbase + (size_t)row * K + kk + colh);
            uint4* db = (uint4*)&tiles[2 + buf][row * LTS + colh];
            db[0] = sb[0]; db[1] = sb[1];
        }
#endif
        __syncthreads();

        const unsigned short* tA = tiles[buf];
        const unsigned short* tB = tiles[2 + buf];
        ABFrag af[4];
#pragma unroll
        for (int mt = 0; mt < 4; ++mt) {
            const unsigned short* p = tA + (wm + mt * 16 + lm) * LTS;
            af[mt].q[0] = *(const uint4*)(p + 16 * hi);       // K 0..7 | 16..23
            af[mt].q[1] = *(const uint4*)(p + 16 * hi + 8);   // K 8..15 | 24..31
        }
        ABFrag bfr[2];
#pragma unroll
        for (int nt = 0; nt < 2; ++nt) {
            const unsigned short* p = tB + (wn + nt * 16 + lm) * LTS + 16 * hi;
            bfr[nt].q[0] = *(const uint4*)(p);
            bfr[nt].q[1] = *(const uint4*)(p + 8);
        }
#pragma unroll
        for (int mt = 0; mt < 4; ++mt)
#pragma unroll
            for (int nt = 0; nt < 2; ++nt)
                acc[mt][nt] = wmma_bf16(af[mt], bfr[nt], acc[mt][nt]);
        __syncthreads();
        buf ^= 1;
    }

#pragma unroll
    for (int mt = 0; mt < 4; ++mt) {
#pragma unroll
        for (int nt = 0; nt < 2; ++nt) {
#pragma unroll
            for (int r = 0; r < 8; ++r) {
                const int row = bm + wm + mt * 16 + hi * 8 + r;
                const int col = bn + wn + nt * 16 + lm;
                const size_t idx = (size_t)row * N + col;
                const float v = acc[mt][nt][r];
                if (EPI == 0)      outF[idx] = v;
                else if (EPI == 1) outF[idx] = v + R[idx];
                else               outB[idx] = f2bf(fmaxf(v, 0.0f));
            }
        }
    }
}

// ---------------------------------------------------------------------------
// QK L2-normalize + layout pack:
//   q,k fp32 [B,S,H,D] -> bf16 [B,H,S,D];  v fp32 [B,S,H,D] -> bf16 [B,H,D,S]
// ---------------------------------------------------------------------------
__global__ __launch_bounds__(256) void pack_qkv_kernel(
    const float* __restrict__ qf, const float* __restrict__ kf,
    const float* __restrict__ vf,
    unsigned short* __restrict__ qh, unsigned short* __restrict__ kh,
    unsigned short* __restrict__ vt)
{
    const int idx = blockIdx.x * 256 + threadIdx.x;
    const int h = idx & 15;
    const int s = (idx >> 4) & 2047;
    const int b = idx >> 15;
    const size_t src  = ((size_t)(b * 2048 + s) * 16 + h) * 64;
    const size_t dsth = ((size_t)(b * 16 + h) * 2048 + s) * 64;

    float ssq = 0.0f;
    for (int d = 0; d < 64; ++d) { const float v = qf[src + d]; ssq += v * v; }
    float inv = rsqrtf(ssq + 1e-6f);
    for (int d = 0; d < 64; ++d) qh[dsth + d] = f2bf(qf[src + d] * inv);

    ssq = 0.0f;
    for (int d = 0; d < 64; ++d) { const float v = kf[src + d]; ssq += v * v; }
    inv = rsqrtf(ssq + 1e-6f);
    for (int d = 0; d < 64; ++d) kh[dsth + d] = f2bf(kf[src + d] * inv);

    const size_t vbase = ((size_t)(b * 16 + h) * 64) * 2048 + s;
    for (int d = 0; d < 64; ++d) vt[vbase + (size_t)d * 2048] = f2bf(vf[src + d]);
}

// ---------------------------------------------------------------------------
// Flash attention with WMMA. One wave per 16-query tile; 32 keys per step.
// T5 bias precomputed into an LDS distance table (kills the per-element
// v_log_f32 chain in the hot loop).
// ---------------------------------------------------------------------------
__global__ __launch_bounds__(256) void attn_kernel(
    const unsigned short* __restrict__ qh, const unsigned short* __restrict__ kh,
    const unsigned short* __restrict__ vt, const float* __restrict__ rel_bias,
    const float* __restrict__ scale, unsigned short* __restrict__ ctx)
{
    const int h = blockIdx.y;
    const int b = blockIdx.z;

    __shared__ float biasH[32];
    __shared__ float biasD[2048];
    __shared__ __align__(16) unsigned short pbuf[8][16 * 40];
    if (threadIdx.x < 32) biasH[threadIdx.x] = rel_bias[threadIdx.x * 16 + h];
    __syncthreads();
    for (int i = threadIdx.x; i < 2048; i += 256) {
        int bk;
        if (i < 16) bk = i;
        else bk = min(31, 16 + (int)(logf((float)i * 0.0625f) * 7.6943526f));
        biasD[i] = biasH[bk];
    }
    __syncthreads();

    const float sc = scale[h];
    const int w    = threadIdx.x >> 5;
    const int lane = threadIdx.x & 31;
    const int lm   = lane & 15;
    const int hi   = lane >> 4;
    const int qb   = blockIdx.x * 128 + w * 16;

    const unsigned short* Qb = qh + (size_t)(b * 16 + h) * 2048 * 64;
    const unsigned short* Kb = kh + (size_t)(b * 16 + h) * 2048 * 64;
    const unsigned short* Vb = vt + (size_t)(b * 16 + h) * 64 * 2048;

    ABFrag qfr[2];
#pragma unroll
    for (int j = 0; j < 2; ++j) {
        const unsigned short* p = Qb + (size_t)(qb + lm) * 64 + 32 * j;
        qfr[j].q[0] = *(const uint4*)(p + 16 * hi);
        qfr[j].q[1] = *(const uint4*)(p + 16 * hi + 8);
    }

    v8f O[4];
#pragma unroll
    for (int t = 0; t < 4; ++t) O[t] = v8f_zero();
    float m_i[8], l_i[8];
#pragma unroll
    for (int r = 0; r < 8; ++r) { m_i[r] = -1.0e6f; l_i[r] = 0.0f; }

    for (int kb = 0; kb <= qb; kb += 32) {
        ABFrag kx[2], ky[2];
#pragma unroll
        for (int j = 0; j < 2; ++j) {
            const unsigned short* p0 = Kb + (size_t)(kb + lm) * 64 + 32 * j + 16 * hi;
            kx[j].q[0] = *(const uint4*)p0;
            kx[j].q[1] = *(const uint4*)(p0 + 8);
            const unsigned short* p1 = Kb + (size_t)(kb + 16 + lm) * 64 + 32 * j + 16 * hi;
            ky[j].q[0] = *(const uint4*)p1;
            ky[j].q[1] = *(const uint4*)(p1 + 8);
        }
        v8f s0 = v8f_zero(), s1 = v8f_zero();
        s0 = wmma_bf16(qfr[0], kx[0], s0);
        s0 = wmma_bf16(qfr[1], kx[1], s0);
        s1 = wmma_bf16(qfr[0], ky[0], s1);
        s1 = wmma_bf16(qfr[1], ky[1], s1);

        float p0e[8], p1e[8];
        const int k0 = kb + lm;
        const int k1 = kb + 16 + lm;
#pragma unroll
        for (int r = 0; r < 8; ++r) {
            const int qpos = qb + hi * 8 + r;
            const int d0 = qpos - k0;
            const int d1 = qpos - k1;
            p0e[r] = (d0 <= 0) ? -1.0e6f : (s0[r] + biasD[d0]) * sc;
            p1e[r] = (d1 <= 0) ? -1.0e6f : (s1[r] + biasD[d1]) * sc;
        }

        float alpha[8];
#pragma unroll
        for (int r = 0; r < 8; ++r) {
            float mx = fmaxf(p0e[r], p1e[r]);
#pragma unroll
            for (int msk = 8; msk >= 1; msk >>= 1)
                mx = fmaxf(mx, __shfl_xor(mx, msk, 32));
            const float mnew = fmaxf(m_i[r], mx);
            alpha[r] = __expf(m_i[r] - mnew);
            m_i[r] = mnew;
            p0e[r] = __expf(p0e[r] - mnew);
            p1e[r] = __expf(p1e[r] - mnew);
            float rs = p0e[r] + p1e[r];
#pragma unroll
            for (int msk = 8; msk >= 1; msk >>= 1)
                rs += __shfl_xor(rs, msk, 32);
            l_i[r] = l_i[r] * alpha[r] + rs;
        }
#pragma unroll
        for (int t = 0; t < 4; ++t)
#pragma unroll
            for (int r = 0; r < 8; ++r) O[t][r] *= alpha[r];

        // Transpose P (C layout -> A layout) through per-wave LDS scratch
#pragma unroll
        for (int r = 0; r < 8; ++r) {
            const int row = hi * 8 + r;
            pbuf[w][row * 40 + lm]      = f2bf(p0e[r]);
            pbuf[w][row * 40 + 16 + lm] = f2bf(p1e[r]);
        }
        __builtin_amdgcn_wave_barrier();   // LDS is in-order within a wave
        ABFrag pf;
        {
            const unsigned short* p = &pbuf[w][lm * 40];
            pf.q[0] = *(const uint4*)(p + 16 * hi);
            pf.q[1] = *(const uint4*)(p + 16 * hi + 8);
        }

#pragma unroll
        for (int t = 0; t < 4; ++t) {
            ABFrag vfr;
            const unsigned short* p = Vb + (size_t)(t * 16 + lm) * 2048 + kb + 16 * hi;
            vfr.q[0] = *(const uint4*)p;
            vfr.q[1] = *(const uint4*)(p + 8);
            O[t] = wmma_bf16(pf, vfr, O[t]);
        }
        __builtin_amdgcn_wave_barrier();
    }

#pragma unroll
    for (int r = 0; r < 8; ++r) {
        const float inv = 1.0f / l_i[r];
        const int qpos = qb + hi * 8 + r;
        const size_t base = ((size_t)(b * 2048 + qpos) * 16 + h) * 64;
#pragma unroll
        for (int t = 0; t < 4; ++t)
            ctx[base + t * 16 + lm] = f2bf(O[t][r] * inv);
    }
}

// ---------------------------------------------------------------------------
// Host-side launch (graph-capture safe: async only, deterministic)
// ---------------------------------------------------------------------------
extern "C" void kernel_launch(void* const* d_in, const int* in_sizes, int n_in,
                              void* d_out, int out_size, void* d_ws, size_t ws_size,
                              hipStream_t stream)
{
    (void)in_sizes; (void)n_in; (void)out_size; (void)ws_size;

    const float* x     = (const float*)d_in[0];
    const float* ln1_w = (const float*)d_in[1];
    const float* Wq    = (const float*)d_in[2];
    const float* Wk    = (const float*)d_in[3];
    const float* Wv    = (const float*)d_in[4];
    const float* rel_b = (const float*)d_in[5];
    const float* scl   = (const float*)d_in[6];
    const float* Wo    = (const float*)d_in[7];
    const float* ln2_w = (const float*)d_in[8];
    const float* W1    = (const float*)d_in[9];
    const float* W2    = (const float*)d_in[10];
    float* out = (float*)d_out;

    char* ws = (char*)d_ws;
    const size_t MB = 1024 * 1024;
    unsigned short* xn  = (unsigned short*)(ws + 0);         //  8 MB bf16 [4096,1024]
    float*          qf_ = (float*)(ws + 8 * MB);             // 16 MB
    float*          kf_ = (float*)(ws + 24 * MB);            // 16 MB
    float*          vf_ = (float*)(ws + 40 * MB);            // 16 MB
    unsigned short* qhd = (unsigned short*)(ws + 56 * MB);   //  8 MB [B,H,S,D]
    unsigned short* khd = (unsigned short*)(ws + 64 * MB);   //  8 MB [B,H,S,D]
    unsigned short* vtd = (unsigned short*)(ws + 72 * MB);   //  8 MB [B,H,D,S]
    unsigned short* ctx = (unsigned short*)(ws + 80 * MB);   //  8 MB [4096,1024]
    float*          y   = (float*)(ws + 88 * MB);            // 16 MB
    unsigned short* zn  = (unsigned short*)(ws + 8 * MB);    // reuse qf_ (dead)
    unsigned short* hb  = (unsigned short*)(ws + 24 * MB);   // reuse kf_+vf_, 32 MB
    unsigned short* wqT = (unsigned short*)(ws + 104 * MB);  //  2 MB bf16 [1024,1024]
    unsigned short* wkT = (unsigned short*)(ws + 106 * MB);
    unsigned short* wvT = (unsigned short*)(ws + 108 * MB);
    unsigned short* woT = (unsigned short*)(ws + 110 * MB);
    unsigned short* w1T = (unsigned short*)(ws + 112 * MB);  //  8 MB bf16 [4096,1024]
    unsigned short* w2T = (unsigned short*)(ws + 120 * MB);  //  8 MB bf16 [1024,4096]

    const int M = 4096;  // B*S
    dim3 blk(256);

    // 0. weight prep: fp32 [K,N] -> bf16 [N,K] (done per call; cheap)
    transpose_bf16_kernel<<<dim3(16, 16), blk, 0, stream>>>(Wq, wqT, 1024, 1024);
    transpose_bf16_kernel<<<dim3(16, 16), blk, 0, stream>>>(Wk, wkT, 1024, 1024);
    transpose_bf16_kernel<<<dim3(16, 16), blk, 0, stream>>>(Wv, wvT, 1024, 1024);
    transpose_bf16_kernel<<<dim3(16, 16), blk, 0, stream>>>(Wo, woT, 1024, 1024);
    transpose_bf16_kernel<<<dim3(64, 16), blk, 0, stream>>>(W1, w1T, 1024, 4096);
    transpose_bf16_kernel<<<dim3(16, 64), blk, 0, stream>>>(W2, w2T, 4096, 1024);

    // 1. pre-attention RMSNorm -> bf16
    rmsnorm_kernel<<<M, blk, 0, stream>>>(x, ln1_w, xn, 1e-6f);

    // 2. Q/K/V projections (WMMA + TDM staging)
    gemm_bf16<0><<<dim3(1024 / BN, M / BM), blk, 0, stream>>>(xn, wqT, qf_, nullptr, nullptr, M, 1024, 1024);
    gemm_bf16<0><<<dim3(1024 / BN, M / BM), blk, 0, stream>>>(xn, wkT, kf_, nullptr, nullptr, M, 1024, 1024);
    gemm_bf16<0><<<dim3(1024 / BN, M / BM), blk, 0, stream>>>(xn, wvT, vf_, nullptr, nullptr, M, 1024, 1024);

    // 3. L2-normalize q,k; pack layouts for attention
    pack_qkv_kernel<<<65536 / 256, blk, 0, stream>>>(qf_, kf_, vf_, qhd, khd, vtd);

    // 4. flash attention (WMMA scores + WMMA P*V)
    attn_kernel<<<dim3(2048 / 128, 16, 2), blk, 0, stream>>>(qhd, khd, vtd, rel_b, scl, ctx);

    // 5. output projection + residual
    gemm_bf16<1><<<dim3(1024 / BN, M / BM), blk, 0, stream>>>(ctx, woT, y, nullptr, x, M, 1024, 1024);

    // 6. pre-FFN RMSNorm
    rmsnorm_kernel<<<M, blk, 0, stream>>>(y, ln2_w, zn, 1e-6f);

    // 7. FFN: relu(zn @ W1) @ W2 + y
    gemm_bf16<2><<<dim3(4096 / BN, M / BM), blk, 0, stream>>>(zn, w1T, nullptr, hb, nullptr, M, 4096, 1024);
    gemm_bf16<1><<<dim3(8, 32), blk, 0, stream>>>(hb, w2T, out, nullptr, y, M, 1024, 4096);
}
